// YOLOXHead_2465311228138
// MI455X (gfx1250) — compile-verified, matrix operand order checked by reading, not run on hardware
//
#include <hip/hip_runtime.h>

typedef __bf16 bf16;
typedef bf16  v16bf __attribute__((ext_vector_type(16)));
typedef bf16  v2bf  __attribute__((ext_vector_type(2)));
typedef bf16  v4bf  __attribute__((ext_vector_type(4)));
typedef float v8f   __attribute__((ext_vector_type(8)));

#define BK 32
#define LDB 40   // LDS row pitch in bf16: 80B rows -> 16B aligned, conflict-free lane stride

__device__ __forceinline__ float sigmoidf_(float x) { return 1.0f / (1.0f + __expf(-x)); }

// ---------------- weight packing: fold BN, f32 -> bf16, layout [oc][tap][cin] ----------------
__global__ void pack_conv(const float* __restrict__ w, const float* __restrict__ g,
                          const float* __restrict__ b, const float* __restrict__ m,
                          const float* __restrict__ v, bf16* __restrict__ wout,
                          float* __restrict__ bout, int cout, int cin, int taps)
{
    int idx = blockIdx.x * blockDim.x + threadIdx.x;
    int tot = cout * cin * taps;
    if (idx < tot) {
        int o = idx / (taps * cin);
        int r = idx % (taps * cin);
        int t = r / cin;
        int i = r % cin;
        float scale = g[o] * rsqrtf(v[o] + 1e-5f);
        wout[((size_t)o * taps + t) * cin + i] = (bf16)(w[(size_t)(o * cin + i) * taps + t] * scale);
    }
    if (idx < cout) {
        float scale = g[idx] * rsqrtf(v[idx] + 1e-5f);
        bout[idx] = b[idx] - m[idx] * scale;
    }
}

// pred weights -> [96][256] bf16: rows 0..3 reg, 4 obj, 5..15 zero, 16..95 cls
__global__ void pack_pred(const float* __restrict__ wcls, const float* __restrict__ bcls,
                          const float* __restrict__ wreg, const float* __restrict__ breg,
                          const float* __restrict__ wobj, const float* __restrict__ bobj,
                          bf16* __restrict__ wout, float* __restrict__ bout)
{
    int idx = blockIdx.x * blockDim.x + threadIdx.x;
    if (idx < 96 * 256) {
        int row = idx >> 8, c = idx & 255;
        float v = 0.0f;
        if (row < 4)        v = wreg[row * 256 + c];
        else if (row == 4)  v = wobj[c];
        else if (row >= 16) v = wcls[(row - 16) * 256 + c];
        wout[idx] = (bf16)v;
    }
    if (idx < 96) {
        float v = 0.0f;
        if (idx < 4)        v = breg[idx];
        else if (idx == 4)  v = bobj[0];
        else if (idx >= 16) v = bcls[idx - 16];
        bout[idx] = v;
    }
}

// ---------------- implicit-GEMM conv + BN(folded) + SiLU, WMMA bf16 ----------------
// block tile: 128 oc x 128 px, 512 threads = 16 waves = (msub 0..7) x (nsub 0..1);
// wave = 16x64 -> 4 WMMA / K-step. LDS double-buffered, software pipelined.
template <typename T>
__global__ __launch_bounds__(512) void conv_bn_silu(
    const T* __restrict__ inp,
    const bf16* __restrict__ wpk, const float* __restrict__ bias,
    bf16* __restrict__ out,
    int cin, int cout, int H, int W, int taps)
{
    const int P = H * W;
    const int n = blockIdx.z;
    const int ocBase = blockIdx.y * 128;
    const int pixBase = blockIdx.x * 128;

    __shared__ __align__(16) bf16 Al[2][128][LDB];
    __shared__ __align__(16) bf16 Bl[2][128][LDB];

    const int tid   = threadIdx.x;
    const int lane  = tid & 31;
    const int wv    = tid >> 5;           // 0..15
    const int msub  = wv & 7;             // 8 x 16 oc
    const int nsub  = wv >> 3;            // 2 x 64 px
    const int rowl  = lane & 15;
    const int khalf = (lane >> 4) << 3;   // 0/8 per ISA 16-bit operand layout

    // A staging: thread -> (oc row 0..127, 8 contiguous k)
    const int am = tid >> 2;
    const int ak = (tid & 3) << 3;
    // B staging: thread -> (4 contiguous pixels, 2 contiguous channels)
    const int pg = (tid & 31) << 2;       // 0..124
    const int cq = (tid >> 5) << 1;       // 0..30
    const int pix0 = pixBase + pg;
    int ph0 = 0, pw0 = 0;
    if (pix0 < P) { ph0 = pix0 / W; pw0 = pix0 % W; }

    const size_t inImg = (size_t)n * cin * P;
    const int cchunks = cin / BK;
    const int steps = taps * cchunks;

    v8f acc[4] = {};
    bf16 aReg[8];
    float bReg[2][4];   // [channel][pixel]

    auto loadChunk = [&](int s) {
        const int t  = s / cchunks;
        const int c0 = (s % cchunks) * BK;
        {   // A: 16B contiguous per thread
            const bf16* srcA = wpk + ((size_t)(ocBase + am) * taps + t) * cin + c0 + ak;
            #pragma unroll
            for (int j = 0; j < 8; ++j) aReg[j] = srcA[j];
        }
        const int dy = (taps == 9) ? (t / 3 - 1) : 0;
        const int dx = (taps == 9) ? (t % 3 - 1) : 0;
        const int y0 = ph0 + dy;
        const int x0 = pw0 + dx;
        const bool fast = (pix0 + 3 < P) && (pw0 + 3 < W) &&
                          (y0 >= 0) && (y0 < H) && (x0 >= 0) && (x0 + 3 < W);
        if (fast) {
            const size_t base = inImg + (size_t)y0 * W + x0;
            #pragma unroll
            for (int c = 0; c < 2; ++c) {
                const T* src = inp + base + (size_t)(c0 + cq + c) * P;
                #pragma unroll
                for (int p = 0; p < 4; ++p) bReg[c][p] = (float)src[p];
            }
        } else {
            #pragma unroll
            for (int p = 0; p < 4; ++p) {
                const int pix = pix0 + p;
                float vs[2] = {0.f, 0.f};
                if (pix < P) {
                    const int hh = pix / W, ww = pix % W;
                    const int yy = hh + dy, xx = ww + dx;
                    if (yy >= 0 && yy < H && xx >= 0 && xx < W) {
                        const size_t sp = inImg + (size_t)yy * W + xx;
                        #pragma unroll
                        for (int c = 0; c < 2; ++c)
                            vs[c] = (float)inp[sp + (size_t)(c0 + cq + c) * P];
                    }
                }
                #pragma unroll
                for (int c = 0; c < 2; ++c) bReg[c][p] = vs[c];
            }
        }
    };

    auto storeChunk = [&](int buf) {
        #pragma unroll
        for (int j = 0; j < 8; ++j) Al[buf][am][ak + j] = aReg[j];
        #pragma unroll
        for (int p = 0; p < 4; ++p) {
            v2bf pk;
            pk[0] = (bf16)bReg[0][p];
            pk[1] = (bf16)bReg[1][p];
            *(v2bf*)&Bl[buf][pg + p][cq] = pk;   // 4B aligned packed store
        }
    };

    auto compute = [&](int buf) {
        v16bf a;
        {
            const bf16* ar = &Al[buf][msub * 16 + rowl][0];
            #pragma unroll
            for (int j = 0; j < 8; ++j) {
                const int k = ((j < 4) ? (2 * j) : (2 * j + 8)) + khalf;
                a[2 * j] = ar[k]; a[2 * j + 1] = ar[k + 1];
            }
        }
        v16bf bf[4];
        #pragma unroll
        for (int s4 = 0; s4 < 4; ++s4) {
            const bf16* br = &Bl[buf][nsub * 64 + s4 * 16 + rowl][0];
            #pragma unroll
            for (int j = 0; j < 8; ++j) {
                const int k = ((j < 4) ? (2 * j) : (2 * j + 8)) + khalf;
                bf[s4][2 * j] = br[k]; bf[s4][2 * j + 1] = br[k + 1];
            }
        }
        #pragma unroll
        for (int s4 = 0; s4 < 4; ++s4)
            acc[s4] = __builtin_amdgcn_wmma_f32_16x16x32_bf16(false, a, false, bf[s4], (short)0, acc[s4], false, false);
    };

    loadChunk(0);
    storeChunk(0);
    __syncthreads();
    for (int s = 0; s < steps; ++s) {
        const int cur = s & 1;
        if (s + 1 < steps) loadChunk(s + 1);   // global loads in flight during WMMA
        compute(cur);
        if (s + 1 < steps) {
            storeChunk(cur ^ 1);               // other buffer: no extra barrier needed
            __syncthreads();
        }
    }

    // epilogue: bias + SiLU, bf16 NCHW store
    #pragma unroll
    for (int r = 0; r < 8; ++r) {
        const int oc = ocBase + msub * 16 + r + khalf;
        const float bz = bias[oc];
        const size_t ob = ((size_t)n * cout + oc) * P;
        #pragma unroll
        for (int s4 = 0; s4 < 4; ++s4) {
            const int p = pixBase + nsub * 64 + s4 * 16 + rowl;
            if (p < P) {
                float v = acc[s4][r] + bz;
                out[ob + p] = (bf16)(v * sigmoidf_(v));
            }
        }
    }
}

// ---------------- pred heads: 96x256 GEMM over 64 px, sigmoid + concat layout ----------------
__global__ __launch_bounds__(192) void pred_head(
    const bf16* __restrict__ cls_feat, const bf16* __restrict__ reg_feat,
    const bf16* __restrict__ wpred, const float* __restrict__ bpred,
    float* __restrict__ out, int H, int W)
{
    const int P = H * W;
    const int n = blockIdx.z;
    const int pixBase = blockIdx.x * 64;
    const int tid   = threadIdx.x;
    const int lane  = tid & 31;
    const int wv    = tid >> 5;          // M tile 0..5
    const int rowl  = lane & 15;
    const int khalf = (lane >> 4) << 3;

    __shared__ __align__(16) bf16 Bc[64][LDB];
    __shared__ __align__(16) bf16 Br[64][LDB];

    v8f acc[4] = {};
    const size_t fBase = (size_t)n * 256 * P;

    for (int c0 = 0; c0 < 256; c0 += BK) {
        for (int idx = tid; idx < 64 * (BK / 4); idx += 192) {   // 512 tasks: 4ch per task
            const int p = idx & 63;
            const int cg = (idx >> 6) << 2;
            const int pix = pixBase + p;
            float vc[4] = {0.f,0.f,0.f,0.f}, vr[4] = {0.f,0.f,0.f,0.f};
            if (pix < P) {
                const size_t gi = fBase + (size_t)(c0 + cg) * P + pix;
                #pragma unroll
                for (int c = 0; c < 4; ++c) { vc[c] = (float)cls_feat[gi + (size_t)c * P];
                                              vr[c] = (float)reg_feat[gi + (size_t)c * P]; }
            }
            v4bf pc, pr;
            #pragma unroll
            for (int c = 0; c < 4; ++c) { pc[c] = (bf16)vc[c]; pr[c] = (bf16)vr[c]; }
            *(v4bf*)&Bc[p][cg] = pc;
            *(v4bf*)&Br[p][cg] = pr;
        }
        __syncthreads();

        v16bf a;
        {
            const bf16* ar = wpred + (size_t)(wv * 16 + rowl) * 256 + c0;
            #pragma unroll
            for (int j = 0; j < 8; ++j) {
                const int k = ((j < 4) ? (2 * j) : (2 * j + 8)) + khalf;
                a[2 * j] = ar[k]; a[2 * j + 1] = ar[k + 1];
            }
        }
        const bf16 (*BB)[LDB] = (wv == 0) ? Br : Bc;
        v16bf bf[4];
        #pragma unroll
        for (int s = 0; s < 4; ++s) {
            const bf16* br = &BB[s * 16 + rowl][0];
            #pragma unroll
            for (int j = 0; j < 8; ++j) {
                const int k = ((j < 4) ? (2 * j) : (2 * j + 8)) + khalf;
                bf[s][2 * j] = br[k]; bf[s][2 * j + 1] = br[k + 1];
            }
        }
        #pragma unroll
        for (int s = 0; s < 4; ++s)
            acc[s] = __builtin_amdgcn_wmma_f32_16x16x32_bf16(false, a, false, bf[s], (short)0, acc[s], false, false);
        __syncthreads();
    }

    #pragma unroll
    for (int r = 0; r < 8; ++r) {
        const int row = wv * 16 + r + khalf;
        int ch; bool sig;
        if (row < 4)       { ch = row;            sig = false; }
        else if (row == 4) { ch = 4;              sig = true;  }
        else if (row < 16) { ch = -1;             sig = false; }
        else               { ch = 5 + (row - 16); sig = true;  }
        if (ch < 0) continue;
        const float bz = bpred[row];
        #pragma unroll
        for (int s = 0; s < 4; ++s) {
            const int pix = pixBase + s * 16 + rowl;
            if (pix >= P) continue;
            float v = acc[s][r] + bz;
            if (sig) v = sigmoidf_(v);
            out[((size_t)n * 85 + ch) * P + pix] = v;
        }
    }
}

// ---------------- host ----------------
extern "C" void kernel_launch(void* const* d_in, const int* in_sizes, int n_in,
                              void* d_out, int out_size, void* d_ws, size_t ws_size,
                              hipStream_t stream)
{
    (void)in_sizes; (void)n_in; (void)out_size; (void)ws_size;
    const int N = 16, HID = 256;
    const int CIN[3] = {256, 512, 1024};
    const int HH[3]  = {80, 40, 20};

    const float* x[3] = {(const float*)d_in[0], (const float*)d_in[1], (const float*)d_in[2]};

    char* ws = (char*)d_ws;
    size_t off = 0;
    auto alloc = [&](size_t bytes) -> char* {
        char* p = ws + off;
        off += (bytes + 255) & ~(size_t)255;
        return p;
    };
    const size_t maxFeatE = (size_t)N * HID * 80 * 80;
    bf16* bufS = (bf16*)alloc(maxFeatE * 2);
    bf16* bufA = (bf16*)alloc(maxFeatE * 2);
    bf16* bufB = (bf16*)alloc(maxFeatE * 2);
    bf16* wStem = (bf16*)alloc((size_t)1024 * 256 * 2);
    bf16* wC0   = (bf16*)alloc((size_t)589824 * 2);
    bf16* wC1   = (bf16*)alloc((size_t)589824 * 2);
    bf16* wR0   = (bf16*)alloc((size_t)589824 * 2);
    bf16* wR1   = (bf16*)alloc((size_t)589824 * 2);
    bf16* wP    = (bf16*)alloc((size_t)96 * 256 * 2);
    float* bStem = (float*)alloc(256 * 4);
    float* bC0   = (float*)alloc(256 * 4);
    float* bC1   = (float*)alloc(256 * 4);
    float* bR0   = (float*)alloc(256 * 4);
    float* bR1   = (float*)alloc(256 * 4);
    float* bP    = (float*)alloc(96 * 4);

    float* outF = (float*)d_out;
    size_t outOff = 0;

    for (int l = 0; l < 3; ++l) {
        // d_in: 0..2 = x0..x2; per level 31 tensors:
        // stem{w,g,b,m,v}=0..4, cls0=5..9, cls1=10..14, reg0=15..19, reg1=20..24,
        // cls_pred{w,b}=25,26, reg_pred=27,28, obj_pred=29,30
        const int base = 3 + l * 31;
        const int cin = CIN[l], H = HH[l], W = HH[l], P = H * W;
        auto IN = [&](int i) { return (const float*)d_in[base + i]; };

        {   // pack all weights for this level (stream-serialized before convs)
            int t1 = HID * cin;
            pack_conv<<<(t1 + 255) / 256, 256, 0, stream>>>(IN(0), IN(1), IN(2), IN(3), IN(4),
                                                            wStem, bStem, HID, cin, 1);
            int t9 = HID * HID * 9;
            pack_conv<<<(t9 + 255) / 256, 256, 0, stream>>>(IN(5), IN(6), IN(7), IN(8), IN(9),
                                                            wC0, bC0, HID, HID, 9);
            pack_conv<<<(t9 + 255) / 256, 256, 0, stream>>>(IN(10), IN(11), IN(12), IN(13), IN(14),
                                                            wC1, bC1, HID, HID, 9);
            pack_conv<<<(t9 + 255) / 256, 256, 0, stream>>>(IN(15), IN(16), IN(17), IN(18), IN(19),
                                                            wR0, bR0, HID, HID, 9);
            pack_conv<<<(t9 + 255) / 256, 256, 0, stream>>>(IN(20), IN(21), IN(22), IN(23), IN(24),
                                                            wR1, bR1, HID, HID, 9);
            pack_pred<<<(96 * 256 + 255) / 256, 256, 0, stream>>>(IN(25), IN(26), IN(27), IN(28),
                                                                  IN(29), IN(30), wP, bP);
        }

        dim3 cgrid((P + 127) / 128, HID / 128, N);
        conv_bn_silu<float><<<cgrid, 512, 0, stream>>>(x[l], wStem, bStem, bufS, cin, HID, H, W, 1);
        conv_bn_silu<bf16 ><<<cgrid, 512, 0, stream>>>(bufS, wC0, bC0, bufA, HID, HID, H, W, 9);
        conv_bn_silu<bf16 ><<<cgrid, 512, 0, stream>>>(bufA, wC1, bC1, bufB, HID, HID, H, W, 9); // cls_feat
        conv_bn_silu<bf16 ><<<cgrid, 512, 0, stream>>>(bufS, wR0, bR0, bufA, HID, HID, H, W, 9);
        conv_bn_silu<bf16 ><<<cgrid, 512, 0, stream>>>(bufA, wR1, bR1, bufS, HID, HID, H, W, 9); // reg_feat

        dim3 pgrid((P + 63) / 64, 1, N);
        pred_head<<<pgrid, 192, 0, stream>>>(bufB, bufS, wP, bP, outF + outOff, H, W);
        outOff += (size_t)N * 85 * P;
    }
}
